// SparseMMDiTBlock_82557861363827
// MI455X (gfx1250) — matrix operand order repaired and checked by reading, not compiled, not run on hardware
//
#include <hip/hip_runtime.h>

typedef unsigned short ushortT;
typedef unsigned int uint32;

typedef __bf16 bf16_t;
typedef bf16_t v16bf __attribute__((ext_vector_type(16)));
typedef float  v8f   __attribute__((ext_vector_type(8)));
typedef uint32 u32x8 __attribute__((ext_vector_type(8)));

// ---- problem constants ----
#define S_VIS   2304
#define S_TXT   512
#define DMODEL  1152
#define NHEADS  16
#define HD      72
#define GRP     576        // S_VIS / 4
#define SEQ     1088       // GRP + S_TXT
#define HDP     96         // hd padded to multiple of 32 (QK K-dim)
#define VR      80         // vT rows (hd padded to multiple of 16)
#define FFD     4608
#define SIXD    (6*DMODEL)

// ---------------- helpers ----------------
__device__ __forceinline__ uint32 f2bfu(float f) {
  uint32 u = __float_as_uint(f);
  return ((u + 0x7fffu + ((u >> 16) & 1u)) >> 16) & 0xffffu;
}
__device__ __forceinline__ float bf2f(ushortT u) {
  return __uint_as_float(((uint32)u) << 16);
}
__device__ __forceinline__ uint32 pack2(float a, float b) {
  return f2bfu(a) | (f2bfu(b) << 16);
}

// A fragment (16-bit 16x32): lane half 0 -> K 0..15, half 1 -> K 16..31 (contiguous 16)
__device__ __forceinline__ v16bf load_afrag(const ushortT* __restrict__ base,
                                            int row, int ld, int k0, int lane) {
  const ushortT* p = base + (size_t)row * ld + k0 + ((lane >> 4) << 4);
  uint4 x0 = *reinterpret_cast<const uint4*>(p);
  uint4 x1 = *reinterpret_cast<const uint4*>(p + 8);
  u32x8 u; u[0]=x0.x; u[1]=x0.y; u[2]=x0.z; u[3]=x0.w;
           u[4]=x1.x; u[5]=x1.y; u[6]=x1.z; u[7]=x1.w;
  return __builtin_bit_cast(v16bf, u);
}
// B fragment (16-bit 32x16, operand stored transposed Bt[N][K]):
// lane half 0 -> K 0..15, half 1 -> K 16..31 (contiguous 16)
__device__ __forceinline__ v16bf load_bfrag(const ushortT* __restrict__ base,
                                            int row, int ld, int k0, int lane) {
  const ushortT* p = base + (size_t)row * ld + k0 + ((lane >> 4) << 4);
  uint4 x0 = *reinterpret_cast<const uint4*>(p);
  uint4 x1 = *reinterpret_cast<const uint4*>(p + 8);
  u32x8 u; u[0]=x0.x; u[1]=x0.y; u[2]=x0.z; u[3]=x0.w;
           u[4]=x1.x; u[5]=x1.y; u[6]=x1.z; u[7]=x1.w;
  return __builtin_bit_cast(v16bf, u);
}

__device__ __forceinline__ v8f wmma_bf16(v16bf a, v16bf b, v8f c) {
  return __builtin_amdgcn_wmma_f32_16x16x32_bf16(false, a, false, b, (short)0, c, false, false);
}

// ---------------- generic bf16 WMMA GEMM ----------------
// C[M,N] = A[M,K](bf16,row-major) * Bt[N,K](bf16) + bias
// OM: 0 -> f32 out, 1 -> bf16 out, 2 -> bf16(gelu_exact(out))
template<int NT, int OM>
__global__ __launch_bounds__(256) void gemm_kernel(const ushortT* __restrict__ A,
                                                   const ushortT* __restrict__ Bt,
                                                   const float* __restrict__ bias,
                                                   void* __restrict__ Cout,
                                                   int M, int N, int K) {
  const int lane = threadIdx.x & 31;
  const int wv   = threadIdx.x >> 5;
  const int m0   = (blockIdx.y * 8 + wv) * 16;
  const int n0   = blockIdx.x * (16 * NT);
  if (m0 >= M) return;
  const v8f vz = {0.f,0.f,0.f,0.f,0.f,0.f,0.f,0.f};
  v8f acc[NT];
#pragma unroll
  for (int t = 0; t < NT; ++t) acc[t] = vz;
  const int l15  = lane & 15;
  const int rowA = m0 + l15;
  for (int k0 = 0; k0 < K; k0 += 32) {
    v16bf a = load_afrag(A, rowA, K, k0, lane);
#pragma unroll
    for (int t = 0; t < NT; ++t) {
      v16bf b = load_bfrag(Bt, n0 + t * 16 + l15, K, k0, lane);
      acc[t] = wmma_bf16(a, b, acc[t]);
    }
  }
  const int half = lane >> 4;
#pragma unroll
  for (int t = 0; t < NT; ++t) {
    const int col = n0 + t * 16 + l15;
    const float bv = bias ? bias[col] : 0.f;
#pragma unroll
    for (int r = 0; r < 8; ++r) {
      const int row = m0 + r + 8 * half;
      float v = acc[t][r] + bv;
      size_t idx = (size_t)row * N + col;
      if (OM == 0)      ((float*)Cout)[idx]   = v;
      else if (OM == 1) ((ushortT*)Cout)[idx] = (ushortT)f2bfu(v);
      else              ((ushortT*)Cout)[idx] =
                          (ushortT)f2bfu(0.5f * v * (1.0f + erff(v * 0.70710678f)));
    }
  }
}

// ---------------- weight cast + transpose (f32 [K,N] -> bf16 [N,K]) ----------------
__global__ __launch_bounds__(256) void castT_kernel(const float* __restrict__ W,
                                                    ushortT* __restrict__ Wt,
                                                    int K, int N) {
  size_t total = (size_t)K * N;
  for (size_t i = (size_t)blockIdx.x * blockDim.x + threadIdx.x; i < total;
       i += (size_t)gridDim.x * blockDim.x) {
    size_t n = i / K, k = i % K;
    Wt[i] = (ushortT)f2bfu(W[k * (size_t)N + n]);
  }
}

// ---------------- modulation ----------------
__global__ void silu_kernel(const float* __restrict__ t, float* __restrict__ o) {
  int i = blockIdx.x * blockDim.x + threadIdx.x;
  if (i < DMODEL) { float x = t[i]; o[i] = x / (1.0f + __expf(-x)); }
}
__global__ __launch_bounds__(256) void modv_kernel(const float* __restrict__ st,
                                                   const float* __restrict__ W,
                                                   const float* __restrict__ b,
                                                   float* __restrict__ out) {
  int j = blockIdx.x * 256 + threadIdx.x;
  if (j >= SIXD) return;
  float acc = b[j];
  for (int t = 0; t < DMODEL; ++t) acc += st[t] * W[(size_t)t * SIXD + j];
  out[j] = acc;
}

// ---------------- RMS norm + adaLN modulation -> bf16 tokens ----------------
__global__ __launch_bounds__(256) void rmsmod_kernel(const float* __restrict__ X,
                                                     const float* __restrict__ w,
                                                     const float* __restrict__ shift,
                                                     const float* __restrict__ scale,
                                                     ushortT* __restrict__ Y, int D) {
  int tok = blockIdx.x, tid = threadIdx.x;
  const float* x = X + (size_t)tok * D;
  __shared__ float red[256];
  float ss = 0.f;
  for (int d = tid; d < D; d += 256) { float v = x[d]; ss += v * v; }
  red[tid] = ss; __syncthreads();
  for (int s = 128; s > 0; s >>= 1) { if (tid < s) red[tid] += red[tid + s]; __syncthreads(); }
  float rs = rsqrtf(red[0] / (float)D + 1e-5f);
  ushortT* y = Y + (size_t)tok * D;
  for (int d = tid; d < D; d += 256)
    y[d] = (ushortT)f2bfu(x[d] * rs * w[d] * (1.0f + scale[d]) + shift[d]);
}

// ---------------- qkv post (visual): head-RMS + RoPE + pack into padded bf16 ----------------
__global__ __launch_bounds__(128) void qkvpost_vis_kernel(const float* __restrict__ Q,
    const float* __restrict__ Kx, const float* __restrict__ V,
    const float* __restrict__ qw, const float* __restrict__ kw,
    const float* __restrict__ cosb, const float* __restrict__ sinb,
    ushortT* __restrict__ qb, ushortT* __restrict__ kb, ushortT* __restrict__ vtb) {
  int s = blockIdx.x, tid = threadIdx.x;
  int ri = s & 3, seq = s >> 2;
  const float* q = Q  + (size_t)s * DMODEL;
  const float* k = Kx + (size_t)s * DMODEL;
  const float* v = V  + (size_t)s * DMODEL;
  __shared__ float hq[16], hk[16], rq[16], rk[16];
  if (tid < 16) { hq[tid] = 0.f; hk[tid] = 0.f; } __syncthreads();
  for (int d = tid; d < DMODEL; d += 128) {
    int h = d / HD;
    atomicAdd(&hq[h], q[d] * q[d]);
    atomicAdd(&hk[h], k[d] * k[d]);
  }
  __syncthreads();
  if (tid < 16) {
    rq[tid] = rsqrtf(hq[tid] * (1.0f / HD) + 1e-5f);
    rk[tid] = rsqrtf(hk[tid] * (1.0f / HD) + 1e-5f);
  }
  __syncthreads();
  for (int d = tid; d < DMODEL; d += 128) {
    int h = d / HD, dd = d - h * HD;
    int pdd = (dd < 36) ? dd + 36 : dd - 36;
    int pd  = h * HD + pdd;
    float c  = cosb[(size_t)s * HD + dd];
    float sn = sinb[(size_t)s * HD + dd];
    float qn = q[d]  * rq[h] * qw[dd];
    float qp = q[pd] * rq[h] * qw[pdd];
    float kn = k[d]  * rk[h] * kw[dd];
    float kp = k[pd] * rk[h] * kw[pdd];
    float qrot = (dd < 36) ? -qp : qp;
    float krot = (dd < 36) ? -kp : kp;
    size_t rh = (size_t)(ri * 16 + h);
    qb[(rh * SEQ + seq) * HDP + dd] = (ushortT)f2bfu(qn * c + qrot * sn);
    kb[(rh * SEQ + seq) * HDP + dd] = (ushortT)f2bfu(kn * c + krot * sn);
    vtb[(rh * VR + dd) * SEQ + seq] = (ushortT)f2bfu(v[d]);
  }
}

// ---------------- qkv post (text): head-RMS, replicate to 4 replicas ----------------
__global__ __launch_bounds__(128) void qkvpost_txt_kernel(const float* __restrict__ Q,
    const float* __restrict__ Kx, const float* __restrict__ V,
    const float* __restrict__ qw, const float* __restrict__ kw,
    ushortT* __restrict__ qb, ushortT* __restrict__ kb, ushortT* __restrict__ vtb) {
  int t = blockIdx.x, tid = threadIdx.x;
  int seq = GRP + t;
  const float* q = Q  + (size_t)t * DMODEL;
  const float* k = Kx + (size_t)t * DMODEL;
  const float* v = V  + (size_t)t * DMODEL;
  __shared__ float hq[16], hk[16], rq[16], rk[16];
  if (tid < 16) { hq[tid] = 0.f; hk[tid] = 0.f; } __syncthreads();
  for (int d = tid; d < DMODEL; d += 128) {
    int h = d / HD;
    atomicAdd(&hq[h], q[d] * q[d]);
    atomicAdd(&hk[h], k[d] * k[d]);
  }
  __syncthreads();
  if (tid < 16) {
    rq[tid] = rsqrtf(hq[tid] * (1.0f / HD) + 1e-5f);
    rk[tid] = rsqrtf(hk[tid] * (1.0f / HD) + 1e-5f);
  }
  __syncthreads();
  for (int d = tid; d < DMODEL; d += 128) {
    int h = d / HD, dd = d - h * HD;
    ushortT qo = (ushortT)f2bfu(q[d] * rq[h] * qw[dd]);
    ushortT ko = (ushortT)f2bfu(k[d] * rk[h] * kw[dd]);
    ushortT vo = (ushortT)f2bfu(v[d]);
#pragma unroll
    for (int r = 0; r < 4; ++r) {
      size_t rh = (size_t)(r * 16 + h);
      qb[(rh * SEQ + seq) * HDP + dd] = qo;
      kb[(rh * SEQ + seq) * HDP + dd] = ko;
      vtb[(rh * VR + dd) * SEQ + seq] = vo;
    }
  }
}

// ---------------- flash attention (transposed-score streaming) ----------------
// grid: (SEQ/64, 64 rh-pairs); block: 128 (4 waves, 16 q-rows each)
__global__ __launch_bounds__(128) void attn_kernel(const ushortT* __restrict__ qb,
                                                   const ushortT* __restrict__ kb,
                                                   const ushortT* __restrict__ vtb,
                                                   float* __restrict__ ao) {
  const int lane = threadIdx.x & 31;
  const int wv   = threadIdx.x >> 5;
  const int rh   = blockIdx.y;
  const int q0   = (blockIdx.x * 4 + wv) * 16;
  const int l15  = lane & 15;
  const bool hi  = lane >= 16;
  const ushortT* qbase = qb  + (size_t)rh * SEQ * HDP;
  const ushortT* kbase = kb  + (size_t)rh * SEQ * HDP;
  const ushortT* vbase = vtb + (size_t)rh * VR * SEQ;

  v16bf bq0 = load_bfrag(qbase, q0 + l15, HDP, 0,  lane);
  v16bf bq1 = load_bfrag(qbase, q0 + l15, HDP, 32, lane);
  v16bf bq2 = load_bfrag(qbase, q0 + l15, HDP, 64, lane);

  const v8f vz = {0.f,0.f,0.f,0.f,0.f,0.f,0.f,0.f};
  v8f o[5];
#pragma unroll
  for (int t = 0; t < 5; ++t) o[t] = vz;
  float mcur = -3.0e38f, lsum = 0.f;
  const float sc = 0.117851130f; // 72^-0.5

  for (int kv0 = 0; kv0 < SEQ; kv0 += 32) {
    v8f s0 = vz, s1 = vz;
    {
      v16bf a;
      a = load_afrag(kbase, kv0 + l15,      HDP, 0,  lane); s0 = wmma_bf16(a, bq0, s0);
      a = load_afrag(kbase, kv0 + l15,      HDP, 32, lane); s0 = wmma_bf16(a, bq1, s0);
      a = load_afrag(kbase, kv0 + l15,      HDP, 64, lane); s0 = wmma_bf16(a, bq2, s0);
      a = load_afrag(kbase, kv0 + 16 + l15, HDP, 0,  lane); s1 = wmma_bf16(a, bq0, s1);
      a = load_afrag(kbase, kv0 + 16 + l15, HDP, 32, lane); s1 = wmma_bf16(a, bq1, s1);
      a = load_afrag(kbase, kv0 + 16 + l15, HDP, 64, lane); s1 = wmma_bf16(a, bq2, s1);
    }
    float p0[8], p1[8];
    float tmax = -3.0e38f;
#pragma unroll
    for (int r = 0; r < 8; ++r) {
      p0[r] = s0[r] * sc; p1[r] = s1[r] * sc;
      tmax = fmaxf(tmax, fmaxf(p0[r], p1[r]));
    }
    tmax = fmaxf(tmax, __shfl_xor(tmax, 16, 32));
    float mnew = fmaxf(mcur, tmax);
    float corr = __expf(mcur - mnew);
    float ps = 0.f;
#pragma unroll
    for (int r = 0; r < 8; ++r) {
      p0[r] = __expf(p0[r] - mnew);
      p1[r] = __expf(p1[r] - mnew);
      ps += p0[r] + p1[r];
    }
    ps += __shfl_xor(ps, 16, 32);
    lsum = lsum * corr + ps;
#pragma unroll
    for (int t = 0; t < 5; ++t)
#pragma unroll
      for (int r = 0; r < 8; ++r) o[t][r] *= corr;
    mcur = mnew;
    // exchange cross-half rows and pack P into B-fragment layout
    float o0[8], o1[8];
#pragma unroll
    for (int r = 0; r < 8; ++r) {
      o0[r] = __shfl_xor(p0[r], 16, 32);
      o1[r] = __shfl_xor(p1[r], 16, 32);
    }
    u32x8 ub;
#pragma unroll
    for (int j = 0; j < 4; ++j) {
      float la = hi ? o1[2*j]   : p0[2*j];
      float lb = hi ? o1[2*j+1] : p0[2*j+1];
      float ha = hi ? p1[2*j]   : o0[2*j];
      float hb = hi ? p1[2*j+1] : o0[2*j+1];
      ub[j]     = pack2(la, lb);
      ub[4 + j] = pack2(ha, hb);
    }
    v16bf bp = __builtin_bit_cast(v16bf, ub);
#pragma unroll
    for (int mt = 0; mt < 5; ++mt) {
      v16bf a = load_afrag(vbase, mt * 16 + l15, SEQ, kv0, lane);
      o[mt] = wmma_bf16(a, bp, o[mt]);
    }
  }
  float inv = 1.0f / lsum;
  const int half = lane >> 4;
#pragma unroll
  for (int mt = 0; mt < 5; ++mt)
#pragma unroll
    for (int r = 0; r < 8; ++r) {
      int d = mt * 16 + r + 8 * half;
      if (d < HD)
        ao[((size_t)(q0 + l15) * 64 + rh) * HD + d] = o[mt][r] * inv;
    }
}

// ---------------- collect attention output back to token layout ----------------
__global__ __launch_bounds__(256) void collect_vis_kernel(const float* __restrict__ ao,
                                                          ushortT* __restrict__ ov) {
  size_t total = (size_t)S_VIS * DMODEL;
  for (size_t i = (size_t)blockIdx.x * blockDim.x + threadIdx.x; i < total;
       i += (size_t)gridDim.x * blockDim.x) {
    int s = (int)(i / DMODEL), d = (int)(i % DMODEL);
    int gi = s >> 2, ri = s & 3, h = d / HD, dd = d - h * HD;
    ov[i] = (ushortT)f2bfu(ao[((size_t)gi * 64 + (ri * 16 + h)) * HD + dd]);
  }
}
__global__ __launch_bounds__(256) void collect_txt_kernel(const float* __restrict__ ao,
                                                          ushortT* __restrict__ ot) {
  size_t total = (size_t)S_TXT * DMODEL;
  for (size_t i = (size_t)blockIdx.x * blockDim.x + threadIdx.x; i < total;
       i += (size_t)gridDim.x * blockDim.x) {
    int t = (int)(i / DMODEL), d = (int)(i % DMODEL);
    int h = d / HD, dd = d - h * HD;
    float acc = 0.f;
#pragma unroll
    for (int r = 0; r < 4; ++r)
      acc += ao[((size_t)(GRP + t) * 64 + (r * 16 + h)) * HD + dd];
    ot[i] = (ushortT)f2bfu(acc * 0.25f);
  }
}

// ---------------- elementwise ----------------
__global__ __launch_bounds__(256) void gated_add_kernel(const float* __restrict__ base,
                                                        const float* __restrict__ gate,
                                                        const float* __restrict__ val,
                                                        float* __restrict__ out,
                                                        size_t n, int D) {
  for (size_t i = (size_t)blockIdx.x * blockDim.x + threadIdx.x; i < n;
       i += (size_t)gridDim.x * blockDim.x)
    out[i] = base[i] + gate[i % D] * val[i];
}
__global__ __launch_bounds__(256) void mulbf_kernel(const ushortT* __restrict__ a,
                                                    const ushortT* __restrict__ b,
                                                    ushortT* __restrict__ o, size_t n) {
  for (size_t i = (size_t)blockIdx.x * blockDim.x + threadIdx.x; i < n;
       i += (size_t)gridDim.x * blockDim.x)
    o[i] = (ushortT)f2bfu(bf2f(a[i]) * bf2f(b[i]));
}

// =======================================================================
extern "C" void kernel_launch(void* const* d_in, const int* in_sizes, int n_in,
                              void* d_out, int out_size, void* d_ws, size_t ws_size,
                              hipStream_t stream) {
  (void)in_sizes; (void)n_in; (void)out_size; (void)ws_size;
  const float* hidden      = (const float*)d_in[0];
  const float* enc         = (const float*)d_in[1];
  const float* temb        = (const float*)d_in[2];
  const float* rope_cos    = (const float*)d_in[3];
  const float* rope_sin    = (const float*)d_in[4];
  const float* w_norm1_vis = (const float*)d_in[5];
  const float* w_norm1_enc = (const float*)d_in[6];
  const float* W_ada1      = (const float*)d_in[7];
  const float* b_ada1      = (const float*)d_in[8];
  const float* Wq          = (const float*)d_in[9];
  const float* Wk          = (const float*)d_in[10];
  const float* Wv          = (const float*)d_in[11];
  const float* q_norm_w    = (const float*)d_in[12];
  const float* k_norm_w    = (const float*)d_in[13];
  const float* Wq_add      = (const float*)d_in[14];
  const float* Wk_add      = (const float*)d_in[15];
  const float* Wv_add      = (const float*)d_in[16];
  const float* addq_norm_w = (const float*)d_in[17];
  const float* addk_norm_w = (const float*)d_in[18];
  const float* Wo          = (const float*)d_in[19];
  const float* bo          = (const float*)d_in[20];
  const float* Wo_add      = (const float*)d_in[21];
  const float* bo_add      = (const float*)d_in[22];
  const float* w_norm2_vis = (const float*)d_in[23];
  const float* w_norm2_enc = (const float*)d_in[24];
  const float* W_ada2      = (const float*)d_in[25];
  const float* b_ada2      = (const float*)d_in[26];
  const float* W1          = (const float*)d_in[27];
  const float* b1          = (const float*)d_in[28];
  const float* W2          = (const float*)d_in[29];
  const float* b2          = (const float*)d_in[30];

  // ---- workspace carving ----
  char* wsp = (char*)d_ws;
  size_t off = 0;
  auto alloc = [&](size_t bytes) -> void* {
    off = (off + 255) & ~(size_t)255;
    void* p = wsp + off;
    off += bytes;
    return p;
  };
  const size_t DD = (size_t)DMODEL * DMODEL;
  float*   stS   = (float*)alloc(DMODEL * 4);
  float*   mod1  = (float*)alloc(SIXD * 4);
  float*   mod2  = (float*)alloc(SIXD * 4);
  ushortT* WqT   = (ushortT*)alloc(DD * 2);
  ushortT* WkT   = (ushortT*)alloc(DD * 2);
  ushortT* WvT   = (ushortT*)alloc(DD * 2);
  ushortT* WqaT  = (ushortT*)alloc(DD * 2);
  ushortT* WkaT  = (ushortT*)alloc(DD * 2);
  ushortT* WvaT  = (ushortT*)alloc(DD * 2);
  ushortT* WoT   = (ushortT*)alloc(DD * 2);
  ushortT* WoaT  = (ushortT*)alloc(DD * 2);
  ushortT* W1T   = (ushortT*)alloc((size_t)2 * FFD * DMODEL * 2);
  ushortT* W2T   = (ushortT*)alloc((size_t)DMODEL * FFD * 2);
  ushortT* hsn   = (ushortT*)alloc((size_t)S_VIS * DMODEL * 2);
  ushortT* enn   = (ushortT*)alloc((size_t)S_TXT * DMODEL * 2);
  float*   qv    = (float*)alloc((size_t)S_VIS * DMODEL * 4);
  float*   kvv   = (float*)alloc((size_t)S_VIS * DMODEL * 4);
  float*   vvv   = (float*)alloc((size_t)S_VIS * DMODEL * 4);
  float*   qtx   = (float*)alloc((size_t)S_TXT * DMODEL * 4);
  float*   ktx   = (float*)alloc((size_t)S_TXT * DMODEL * 4);
  float*   vtx   = (float*)alloc((size_t)S_TXT * DMODEL * 4);
  const size_t qkbytes = (size_t)64 * SEQ * HDP * 2;
  const size_t vtbytes = (size_t)64 * VR * SEQ * 2;
  ushortT* qbuf  = (ushortT*)alloc(qkbytes);
  ushortT* kbuf  = (ushortT*)alloc(qkbytes);
  ushortT* vtb   = (ushortT*)alloc(vtbytes);
  float*   ao    = (float*)alloc((size_t)SEQ * 64 * HD * 4);
  ushortT* ovb   = (ushortT*)alloc((size_t)S_VIS * DMODEL * 2);
  ushortT* otb   = (ushortT*)alloc((size_t)S_TXT * DMODEL * 2);
  float*   avis  = (float*)alloc((size_t)S_VIS * DMODEL * 4);
  float*   atxt  = (float*)alloc((size_t)S_TXT * DMODEL * 4);
  float*   hid2  = (float*)alloc((size_t)S_VIS * DMODEL * 4);
  float*   enc2  = (float*)alloc((size_t)S_TXT * DMODEL * 4);
  const int STOT = S_VIS + S_TXT;  // 2816
  ushortT* xn    = (ushortT*)alloc((size_t)STOT * DMODEL * 2);
  ushortT* xhB   = (ushortT*)alloc((size_t)STOT * FFD * 2);
  ushortT* gxgB  = (ushortT*)alloc((size_t)STOT * FFD * 2);
  ushortT* hg    = (ushortT*)alloc((size_t)STOT * FFD * 2);
  float*   ffb   = (float*)alloc((size_t)STOT * DMODEL * 4);
  float*   out   = (float*)d_out;

  // ---- 1. weight cast + transpose ----
  castT_kernel<<<4096, 256, 0, stream>>>(Wq, WqT, DMODEL, DMODEL);
  castT_kernel<<<4096, 256, 0, stream>>>(Wk, WkT, DMODEL, DMODEL);
  castT_kernel<<<4096, 256, 0, stream>>>(Wv, WvT, DMODEL, DMODEL);
  castT_kernel<<<4096, 256, 0, stream>>>(Wq_add, WqaT, DMODEL, DMODEL);
  castT_kernel<<<4096, 256, 0, stream>>>(Wk_add, WkaT, DMODEL, DMODEL);
  castT_kernel<<<4096, 256, 0, stream>>>(Wv_add, WvaT, DMODEL, DMODEL);
  castT_kernel<<<4096, 256, 0, stream>>>(Wo, WoT, DMODEL, DMODEL);
  castT_kernel<<<4096, 256, 0, stream>>>(Wo_add, WoaT, DMODEL, DMODEL);
  castT_kernel<<<4096, 256, 0, stream>>>(W1, W1T, DMODEL, 2 * FFD);
  castT_kernel<<<4096, 256, 0, stream>>>(W2, W2T, FFD, DMODEL);

  // ---- 2. modulation ----
  silu_kernel<<<(DMODEL + 255) / 256, 256, 0, stream>>>(temb, stS);
  modv_kernel<<<(SIXD + 255) / 256, 256, 0, stream>>>(stS, W_ada1, b_ada1, mod1);
  modv_kernel<<<(SIXD + 255) / 256, 256, 0, stream>>>(stS, W_ada2, b_ada2, mod2);

  // ---- 3. norm1 + modulation ----
  rmsmod_kernel<<<S_VIS, 256, 0, stream>>>(hidden, w_norm1_vis, mod1, mod1 + DMODEL, hsn, DMODEL);
  rmsmod_kernel<<<S_TXT, 256, 0, stream>>>(enc, w_norm1_enc, mod1 + 3 * DMODEL, mod1 + 4 * DMODEL, enn, DMODEL);

  // ---- 4. QKV projections ----
  dim3 gV(DMODEL / 64, S_VIS / 128), gT(DMODEL / 64, S_TXT / 128);
  gemm_kernel<4, 0><<<gV, 256, 0, stream>>>(hsn, WqT, nullptr, qv,  S_VIS, DMODEL, DMODEL);
  gemm_kernel<4, 0><<<gV, 256, 0, stream>>>(hsn, WkT, nullptr, kvv, S_VIS, DMODEL, DMODEL);
  gemm_kernel<4, 0><<<gV, 256, 0, stream>>>(hsn, WvT, nullptr, vvv, S_VIS, DMODEL, DMODEL);
  gemm_kernel<4, 0><<<gT, 256, 0, stream>>>(enn, WqaT, nullptr, qtx, S_TXT, DMODEL, DMODEL);
  gemm_kernel<4, 0><<<gT, 256, 0, stream>>>(enn, WkaT, nullptr, ktx, S_TXT, DMODEL, DMODEL);
  gemm_kernel<4, 0><<<gT, 256, 0, stream>>>(enn, WvaT, nullptr, vtx, S_TXT, DMODEL, DMODEL);

  // ---- 5. zero the padded attention operand buffers ----
  hipMemsetAsync(qbuf, 0, qkbytes, stream);
  hipMemsetAsync(kbuf, 0, qkbytes, stream);
  hipMemsetAsync(vtb,  0, vtbytes, stream);

  // ---- 6. per-head RMS + RoPE + pack ----
  qkvpost_vis_kernel<<<S_VIS, 128, 0, stream>>>(qv, kvv, vvv, q_norm_w, k_norm_w,
                                                rope_cos, rope_sin, qbuf, kbuf, vtb);
  qkvpost_txt_kernel<<<S_TXT, 128, 0, stream>>>(qtx, ktx, vtx, addq_norm_w, addk_norm_w,
                                                qbuf, kbuf, vtb);

  // ---- 7. attention ----
  attn_kernel<<<dim3(SEQ / 64, 64), 128, 0, stream>>>(qbuf, kbuf, vtb, ao);

  // ---- 8. collect + output projections ----
  collect_vis_kernel<<<4096, 256, 0, stream>>>(ao, ovb);
  collect_txt_kernel<<<2304, 256, 0, stream>>>(ao, otb);
  gemm_kernel<4, 0><<<gV, 256, 0, stream>>>(ovb, WoT, bo, avis, S_VIS, DMODEL, DMODEL);
  gemm_kernel<4, 0><<<gT, 256, 0, stream>>>(otb, WoaT, bo_add, atxt, S_TXT, DMODEL, DMODEL);

  // ---- 9. gated residual #1 ----
  gated_add_kernel<<<4096, 256, 0, stream>>>(hidden, mod1 + 2 * DMODEL, avis, hid2,
                                             (size_t)S_VIS * DMODEL, DMODEL);
  gated_add_kernel<<<2304, 256, 0, stream>>>(enc, mod1 + 5 * DMODEL, atxt, enc2,
                                             (size_t)S_TXT * DMODEL, DMODEL);

  // ---- 10. norm2 + modulation (concat into xn) ----
  rmsmod_kernel<<<S_VIS, 256, 0, stream>>>(hid2, w_norm2_vis, mod2, mod2 + DMODEL, xn, DMODEL);
  rmsmod_kernel<<<S_TXT, 256, 0, stream>>>(enc2, w_norm2_enc, mod2 + 3 * DMODEL, mod2 + 4 * DMODEL,
                                           xn + (size_t)S_VIS * DMODEL, DMODEL);

  // ---- 11. GEGLU FF ----
  dim3 gF1(FFD / 64, STOT / 128), gF2(DMODEL / 64, STOT / 128);
  gemm_kernel<4, 1><<<gF1, 256, 0, stream>>>(xn, W1T, b1, xhB, STOT, FFD, DMODEL);
  gemm_kernel<4, 2><<<gF1, 256, 0, stream>>>(xn, W1T + (size_t)FFD * DMODEL, b1 + FFD,
                                             gxgB, STOT, FFD, DMODEL);
  mulbf_kernel<<<4096, 256, 0, stream>>>(xhB, gxgB, hg, (size_t)STOT * FFD);
  gemm_kernel<4, 0><<<gF2, 256, 0, stream>>>(hg, W2T, b2, ffb, STOT, DMODEL, FFD);

  // ---- 12. gated residual #2 -> d_out ----
  gated_add_kernel<<<4096, 256, 0, stream>>>(hid2, mod2 + 2 * DMODEL, ffb, out,
                                             (size_t)S_VIS * DMODEL, DMODEL);
  gated_add_kernel<<<2304, 256, 0, stream>>>(enc2, mod2 + 5 * DMODEL,
                                             ffb + (size_t)S_VIS * DMODEL,
                                             out + (size_t)S_VIS * DMODEL,
                                             (size_t)S_TXT * DMODEL, DMODEL);
}
